// MPNN_48404281426498
// MI455X (gfx1250) — compile-verified, matrix operand order checked by reading, not run on hardware
//
#include <hip/hip_runtime.h>
#include <hip/hip_bf16.h>

// ---------------- problem constants (match reference) ----------------
#define N_ATOMS   300000
#define N_BONDS   600000
#define MAX_NB    6
#define ATOM_FDIM 133
#define BOND_FDIM 147
#define HIDDEN    256
#define DEPTH     3
#define N_MOLS    10000

#define KPAD      160          // 147/133 padded to 5 chunks of 32

typedef __attribute__((ext_vector_type(16))) _Float16 v16h;
typedef __attribute__((ext_vector_type(8)))  _Float16 v8h;
typedef __attribute__((ext_vector_type(8)))  float    v8f;

// -------------------------------------------------------------------
// Fragment layouts (ISA 7.12.2), wave32, lane = hi*16 + lo:
//  A (16x32 MxK): row M = lo; halves 0..7  = K=k0+hi*8+{0..7},
//                              halves 8..15 = K=k0+16+hi*8+{0..7}
//  B (32x16 KxN): symmetric, lane holds column N = lo, same K split.
//  C/D (16x16):   lane holds column N = lo; VGPR v = row M = hi*8+v.
// B is pre-swizzled so a lane's 16 halves are one contiguous 32B load:
//  halves at out[((c*256 + n)*2 + hi)*16 + i]
// -------------------------------------------------------------------

// ---------------- weight pre-swizzle: f32 [ksrc,256] -> frag-order f16 ----
__global__ __launch_bounds__(256)
void mpnn_prep_w(const float* __restrict__ W, _Float16* __restrict__ out,
                 int ksrc, int kchunks) {
  int idx = blockIdx.x * 256 + threadIdx.x;       // one output half each
  int total = kchunks * 256 * 32;
  if (idx >= total) return;
  int i  = idx & 15;
  int hi = (idx >> 4) & 1;
  int n  = (idx >> 5) & 255;
  int c  = idx >> 13;
  int k  = c * 32 + hi * 8 + (i < 8 ? i : i + 8);
  float v = (k < ksrc) ? W[(long)k * HIDDEN + n] : 0.0f;
  out[idx] = (_Float16)v;
}

// ---------------- activation pad+convert: f32 [rows,ksrc] -> f16 [rows,kpad]
__global__ __launch_bounds__(256)
void mpnn_prep_a(const float* __restrict__ A, _Float16* __restrict__ out,
                 int rows, int ksrc, int kpad) {
  long idx = (long)blockIdx.x * 256 + threadIdx.x;   // one per 4 halves
  int k4 = kpad >> 2;
  long total = (long)rows * k4;
  if (idx >= total) return;
  int  g   = (int)(idx % k4);
  long row = idx / k4;
  const float* src = A + row * ksrc;
  _Float16* dst = out + row * kpad + g * 4;
#pragma unroll
  for (int j = 0; j < 4; ++j) {
    int k = g * 4 + j;
    dst[j] = (k < ksrc) ? (_Float16)src[k] : (_Float16)0.0f;
  }
}

// ---------------- shared WMMA inner loop: 16 rows x 64 cols, K=KC*32 ------
template <int LDA, int KC>
__device__ __forceinline__ void gemm_acc(const _Float16* __restrict__ Abase,
                                         const _Float16* __restrict__ Bsw,
                                         int m_base, int n_base, int lo, int hi,
                                         v8f acc[4]) {
  const _Float16* arow = Abase + (long)(m_base + lo) * LDA;
#pragma unroll
  for (int c = 0; c < KC; ++c) {
    const int ka = c * 32 + hi * 8;
    v8h p = *(const v8h*)(arow + ka);        // 16B aligned
    v8h q = *(const v8h*)(arow + ka + 16);   // 16B aligned
    v16h a = __builtin_shufflevector(p, q, 0, 1, 2, 3, 4, 5, 6, 7,
                                     8, 9, 10, 11, 12, 13, 14, 15);
#pragma unroll
    for (int t = 0; t < 4; ++t) {
      const _Float16* bp =
          Bsw + (((long)c * 256 + n_base + t * 16 + lo) * 2 + hi) * 16;
      v16h b = *(const v16h*)bp;             // 32B aligned
      acc[t] = __builtin_amdgcn_wmma_f32_16x16x32_f16(
          false, a, false, b, (short)0, acc[t], false, false);
    }
  }
}

// ---------------- K1: inp16 = fb16 @ W_i ; msg16 = relu(inp16) ------------
__global__ __launch_bounds__(256)
void mpnn_gemm_wi(const _Float16* __restrict__ fb16,
                  const _Float16* __restrict__ Wi16,
                  _Float16* __restrict__ inp16, _Float16* __restrict__ msg16) {
  const int wave = blockIdx.x * 8 + (threadIdx.x >> 5);
  const int m_base = (wave >> 2) * 16;
  const int n_base = (wave & 3) * 64;
  const int lane = threadIdx.x & 31;
  const int lo = lane & 15, hi = lane >> 4;

  v8f z = {};
  v8f acc[4] = {z, z, z, z};
  gemm_acc<KPAD, KPAD / 32>(fb16, Wi16, m_base, n_base, lo, hi, acc);

#pragma unroll
  for (int t = 0; t < 4; ++t) {
    int n = n_base + t * 16 + lo;
#pragma unroll
    for (int v = 0; v < 8; ++v) {
      long off = (long)(m_base + hi * 8 + v) * HIDDEN + n;
      float val = acc[t][v];
      inp16[off] = (_Float16)val;
      msg16[off] = (_Float16)fmaxf(val, 0.0f);
    }
  }
}

// ---------------- K3: msg16 = relu(inp16 + tbuf16 @ W_h) ------------------
__global__ __launch_bounds__(256)
void mpnn_gemm_wh(const _Float16* __restrict__ tbuf16,
                  const _Float16* __restrict__ Wh16,
                  const _Float16* __restrict__ inp16,
                  _Float16* __restrict__ msg16) {
  const int wave = blockIdx.x * 8 + (threadIdx.x >> 5);
  const int m_base = (wave >> 2) * 16;
  const int n_base = (wave & 3) * 64;
  const int lane = threadIdx.x & 31;
  const int lo = lane & 15, hi = lane >> 4;

  v8f z = {};
  v8f acc[4] = {z, z, z, z};
  gemm_acc<HIDDEN, HIDDEN / 32>(tbuf16, Wh16, m_base, n_base, lo, hi, acc);

#pragma unroll
  for (int t = 0; t < 4; ++t) {
    int n = n_base + t * 16 + lo;
#pragma unroll
    for (int v = 0; v < 8; ++v) {
      long off = (long)(m_base + hi * 8 + v) * HIDDEN + n;
      float val = acc[t][v] + (float)inp16[off];
      msg16[off] = (_Float16)fmaxf(val, 0.0f);
    }
  }
}

// ---------------- K4: relu([f_atoms|amsg] @ W_o + b_o) -> atomic segsum ---
__global__ __launch_bounds__(256)
void mpnn_gemm_wo(const _Float16* __restrict__ fa16,
                  const _Float16* __restrict__ amsg16,
                  const _Float16* __restrict__ Wo16a,
                  const _Float16* __restrict__ Wo16b,
                  const float* __restrict__ b_o, const int* __restrict__ seg,
                  float* __restrict__ molsum) {
  const int wave = blockIdx.x * 8 + (threadIdx.x >> 5);
  const int m_base = (wave >> 2) * 16;
  const int n_base = (wave & 3) * 64;
  const int lane = threadIdx.x & 31;
  const int lo = lane & 15, hi = lane >> 4;

  v8f z = {};
  v8f acc[4] = {z, z, z, z};
  gemm_acc<KPAD, KPAD / 32>(fa16, Wo16a, m_base, n_base, lo, hi, acc);
  gemm_acc<HIDDEN, HIDDEN / 32>(amsg16, Wo16b, m_base, n_base, lo, hi, acc);

  int s[8];
#pragma unroll
  for (int v = 0; v < 8; ++v) s[v] = seg[m_base + hi * 8 + v];

#pragma unroll
  for (int t = 0; t < 4; ++t) {
    int n = n_base + t * 16 + lo;
    float bias = b_o[n];
#pragma unroll
    for (int v = 0; v < 8; ++v) {
      float val = fmaxf(acc[t][v] + bias, 0.0f);
      atomicAdd(&molsum[(long)s[v] * HIDDEN + n], val);
    }
  }
}

// ---------------- gathers (f16 rows, 16B vectors, f32 accumulate) ---------
__global__ __launch_bounds__(256)
void mpnn_gather_sum(const v8h* __restrict__ msg, const int* __restrict__ a2b,
                     v8h* __restrict__ amsg) {
  int idx = blockIdx.x * 256 + threadIdx.x;
  int atom = idx >> 5;                    // 32 v8h per 256-half row
  int q = idx & 31;
  if (atom >= N_ATOMS) return;
  float s[8] = {0.f, 0.f, 0.f, 0.f, 0.f, 0.f, 0.f, 0.f};
#pragma unroll
  for (int j = 0; j < MAX_NB; ++j) {
    int b = a2b[atom * MAX_NB + j];
    v8h m = msg[(long)b * 32 + q];
#pragma unroll
    for (int e = 0; e < 8; ++e) s[e] += (float)m[e];
  }
  v8h r;
#pragma unroll
  for (int e = 0; e < 8; ++e) r[e] = (_Float16)s[e];
  amsg[(long)atom * 32 + q] = r;
}

__global__ __launch_bounds__(256)
void mpnn_bond_gather(const v8h* __restrict__ amsg, const v8h* __restrict__ msg,
                      const int* __restrict__ b2a,
                      const int* __restrict__ b2revb,
                      v8h* __restrict__ tbuf) {
  int idx = blockIdx.x * 256 + threadIdx.x;
  int bond = idx >> 5;
  int q = idx & 31;
  if (bond >= N_BONDS) return;
  v8h x = amsg[(long)b2a[bond] * 32 + q];
  v8h y = msg[(long)b2revb[bond] * 32 + q];
  v8h r;
#pragma unroll
  for (int e = 0; e < 8; ++e) r[e] = (_Float16)((float)x[e] - (float)y[e]);
  tbuf[(long)bond * 32 + q] = r;
}

// ---------------- zero / counts / head ------------------------------------
__global__ __launch_bounds__(256)
void mpnn_zero(float* __restrict__ p, int n) {
  int i = blockIdx.x * 256 + threadIdx.x;
  if (i < n) p[i] = 0.0f;
}

__global__ __launch_bounds__(256)
void mpnn_counts(const int* __restrict__ seg, float* __restrict__ counts) {
  int i = blockIdx.x * 256 + threadIdx.x;
  if (i < N_ATOMS) atomicAdd(&counts[seg[i]], 1.0f);
}

__global__ __launch_bounds__(256)
void mpnn_head(const float* __restrict__ molsum, const float* __restrict__ counts,
               const float* __restrict__ W_ffn, const float* __restrict__ b_ffn,
               float* __restrict__ out) {
  __shared__ float red[256];
  int mol = blockIdx.x;
  int h = threadIdx.x;
  float inv = 1.0f / fmaxf(counts[mol], 1.0f);
  red[h] = molsum[(long)mol * HIDDEN + h] * inv * W_ffn[h];
  __syncthreads();
  for (int sft = 128; sft > 0; sft >>= 1) {
    if (h < sft) red[h] += red[h + sft];
    __syncthreads();
  }
  if (h == 0) out[mol] = red[0] + b_ffn[0];
}

// ---------------- launch --------------------------------------------------
extern "C" void kernel_launch(void* const* d_in, const int* in_sizes, int n_in,
                              void* d_out, int out_size, void* d_ws,
                              size_t ws_size, hipStream_t stream) {
  const float* f_atoms = (const float*)d_in[0];
  const float* f_bonds = (const float*)d_in[1];
  const float* W_i     = (const float*)d_in[2];
  const float* W_h     = (const float*)d_in[3];
  const float* W_o     = (const float*)d_in[4];
  const float* b_o     = (const float*)d_in[5];
  const float* W_ffn   = (const float*)d_in[6];
  const float* b_ffn   = (const float*)d_in[7];
  const int*   a2b     = (const int*)d_in[8];
  const int*   b2a     = (const int*)d_in[9];
  const int*   b2revb  = (const int*)d_in[10];
  const int*   seg     = (const int*)d_in[11];
  float* out = (float*)d_out;

  // ---- workspace carve-up (256B aligned) ----
  char* p = (char*)d_ws;
  auto alloc = [&](size_t bytes) {
    char* r = p;
    p += (bytes + 255) & ~(size_t)255;
    return r;
  };
  const size_t HB = (size_t)N_BONDS * HIDDEN;   // halves per bond tensor
  const size_t HA = (size_t)N_ATOMS * HIDDEN;
  _Float16* inp16  = (_Float16*)alloc(HB * 2);
  _Float16* msg16  = (_Float16*)alloc(HB * 2);
  _Float16* tbuf16 = (_Float16*)alloc(HB * 2);
  _Float16* amsg16 = (_Float16*)alloc(HA * 2);
  _Float16* fb16   = (_Float16*)alloc((size_t)N_BONDS * KPAD * 2);
  _Float16* fa16   = (_Float16*)alloc((size_t)N_ATOMS * KPAD * 2);
  _Float16* Wi16   = (_Float16*)alloc((size_t)KPAD * HIDDEN * 2);
  _Float16* Wh16   = (_Float16*)alloc((size_t)HIDDEN * HIDDEN * 2);
  _Float16* Wo16a  = (_Float16*)alloc((size_t)KPAD * HIDDEN * 2);
  _Float16* Wo16b  = (_Float16*)alloc((size_t)HIDDEN * HIDDEN * 2);
  float*    molsum = (float*)alloc((size_t)N_MOLS * HIDDEN * 4 + N_MOLS * 4);
  float*    counts = molsum + (size_t)N_MOLS * HIDDEN;

  // ---- pre-passes: weight swizzle + activation pad/convert ----
  mpnn_prep_w<<<(5 * 8192 + 255) / 256, 256, 0, stream>>>(W_i, Wi16, BOND_FDIM, 5);
  mpnn_prep_w<<<(8 * 8192 + 255) / 256, 256, 0, stream>>>(W_h, Wh16, HIDDEN, 8);
  mpnn_prep_w<<<(5 * 8192 + 255) / 256, 256, 0, stream>>>(W_o, Wo16a, ATOM_FDIM, 5);
  mpnn_prep_w<<<(8 * 8192 + 255) / 256, 256, 0, stream>>>(
      W_o + (size_t)ATOM_FDIM * HIDDEN, Wo16b, HIDDEN, 8);
  {
    long tb = (long)N_BONDS * (KPAD / 4);
    long ta = (long)N_ATOMS * (KPAD / 4);
    mpnn_prep_a<<<(int)((tb + 255) / 256), 256, 0, stream>>>(
        f_bonds, fb16, N_BONDS, BOND_FDIM, KPAD);
    mpnn_prep_a<<<(int)((ta + 255) / 256), 256, 0, stream>>>(
        f_atoms, fa16, N_ATOMS, ATOM_FDIM, KPAD);
  }

  const int gemm_bond_blocks = (N_BONDS / 16) * 4 / 8;  // 18750
  const int gemm_atom_blocks = (N_ATOMS / 16) * 4 / 8;  //  9375
  const int gsum_blocks = N_ATOMS * 32 / 256;           // 37500
  const int bgat_blocks = N_BONDS * 32 / 256;           // 75000

  // ---- message passing ----
  mpnn_gemm_wi<<<gemm_bond_blocks, 256, 0, stream>>>(fb16, Wi16, inp16, msg16);

  for (int d = 0; d < DEPTH - 1; ++d) {
    mpnn_gather_sum<<<gsum_blocks, 256, 0, stream>>>(
        (const v8h*)msg16, a2b, (v8h*)amsg16);
    mpnn_bond_gather<<<bgat_blocks, 256, 0, stream>>>(
        (const v8h*)amsg16, (const v8h*)msg16, b2a, b2revb, (v8h*)tbuf16);
    mpnn_gemm_wh<<<gemm_bond_blocks, 256, 0, stream>>>(tbuf16, Wh16, inp16,
                                                       msg16);
  }
  mpnn_gather_sum<<<gsum_blocks, 256, 0, stream>>>(
      (const v8h*)msg16, a2b, (v8h*)amsg16);

  // ---- readout ----
  const int zn = N_MOLS * HIDDEN + N_MOLS;
  mpnn_zero<<<(zn + 255) / 256, 256, 0, stream>>>(molsum, zn);
  mpnn_counts<<<(N_ATOMS + 255) / 256, 256, 0, stream>>>(seg, counts);
  mpnn_gemm_wo<<<gemm_atom_blocks, 256, 0, stream>>>(fa16, amsg16, Wo16a, Wo16b,
                                                     b_o, seg, molsum);
  mpnn_head<<<N_MOLS, 256, 0, stream>>>(molsum, counts, W_ffn, b_ffn, out);
}
